// TransformerBlock_4939212391075
// MI455X (gfx1250) — matrix operand order repaired and checked by reading, not compiled, not run on hardware
//
#include <hip/hip_runtime.h>
#include <hip/hip_bf16.h>

// ---------------------------------------------------------------------------
// Problem constants (from reference)
// ---------------------------------------------------------------------------
#define Nn    10000
#define Ee    160000
#define Dd    128
#define Hh    4
#define CC    128
#define HCc   512
#define LDN   1664          // q(512) | k(512) | v(512) | skip(128)
#define MTILES 626          // 10000/16 + 1; last tile is all padding
#define NTILES 104          // 1664/16
#define MPAD  10016         // MTILES*16
#define KCHUNK (NTILES * 32 * 16)   // 53248 shorts between kt-chunks in packed B
#define NTSTRIDE (32 * 16)          // 512 shorts between consecutive n-tiles

typedef __attribute__((ext_vector_type(16))) __bf16 v16bf;
typedef __attribute__((ext_vector_type(8)))  float  v8f;

// ---------------------------------------------------------------------------
// Helpers
// ---------------------------------------------------------------------------
static __device__ __forceinline__ unsigned short f2bf(float f) {
    unsigned int u = __float_as_uint(f);
    unsigned int r = u + 0x7FFFu + ((u >> 16) & 1u);   // round-to-nearest-even
    return (unsigned short)(r >> 16);
}
// monotone float <-> uint mapping for atomicMax-based segment max
static __device__ __forceinline__ unsigned int f2ord(float f) {
    unsigned int u = __float_as_uint(f);
    return (u & 0x80000000u) ? ~u : (u | 0x80000000u);
}
static __device__ __forceinline__ float ord2f(unsigned int u) {
    unsigned int b = (u & 0x80000000u) ? (u & 0x7FFFFFFFu) : ~u;
    return __uint_as_float(b);
}

struct BTile { v16bf b0, b1, b2, b3; };
static __device__ __forceinline__ BTile load_btile(const unsigned short* __restrict__ p) {
    BTile t;
    ((uint4*)&t.b0)[0] = *(const uint4*)(p);
    ((uint4*)&t.b0)[1] = *(const uint4*)(p + 8);
    ((uint4*)&t.b1)[0] = *(const uint4*)(p + KCHUNK);
    ((uint4*)&t.b1)[1] = *(const uint4*)(p + KCHUNK + 8);
    ((uint4*)&t.b2)[0] = *(const uint4*)(p + 2 * KCHUNK);
    ((uint4*)&t.b2)[1] = *(const uint4*)(p + 2 * KCHUNK + 8);
    ((uint4*)&t.b3)[0] = *(const uint4*)(p + 3 * KCHUNK);
    ((uint4*)&t.b3)[1] = *(const uint4*)(p + 3 * KCHUNK + 8);
    return t;
}

// ---------------------------------------------------------------------------
// Pack kernels: f32 -> bf16, pre-swizzled into WMMA lane layouts
// ---------------------------------------------------------------------------
__global__ void pack_x(const float* __restrict__ x, unsigned short* __restrict__ Xb) {
    int i = blockIdx.x * 256 + threadIdx.x;
    if (i >= MPAD * Dd) return;
    int row = i >> 7;
    Xb[i] = (row < Nn) ? f2bf(x[i]) : (unsigned short)0;
}

// B packed layout: chunk ((kt*NTILES+nt)*32 + lane)*16 holds the 16 contiguous
// bf16 this lane feeds into v_wmma (lane 0-15: N=lane, K=kt*32+{0..7,16..23};
// lane 16-31: N=lane-16, K=kt*32+{8..15,24..31})
__global__ void pack_b(const float* __restrict__ Wq, const float* __restrict__ Wk,
                       const float* __restrict__ Wv, const float* __restrict__ Wskip,
                       unsigned short* __restrict__ Bp) {
    int tid = blockIdx.x * 256 + threadIdx.x;
    if (tid >= 4 * NTILES * 32 * 16) return;
    int i    = tid & 15;
    int lane = (tid >> 4) & 31;
    int t    = tid >> 9;
    int nt   = t % NTILES;
    int kt   = t / NTILES;
    int hi   = lane >> 4;
    int n    = nt * 16 + (lane & 15);
    int k    = kt * 32 + hi * 8 + (i & 7) + (i >> 3) * 16;
    float w;
    if      (n < 512)  w = Wq[k * HCc + n];
    else if (n < 1024) w = Wk[k * HCc + (n - 512)];
    else if (n < 1536) w = Wv[k * HCc + (n - 1024)];
    else               w = Wskip[k * Dd + (n - 1536)];
    Bp[tid] = f2bf(w);
}

__global__ void pack_bias(const float* __restrict__ bq, const float* __restrict__ bk,
                          const float* __restrict__ bv, const float* __restrict__ bskip,
                          float* __restrict__ biasc) {
    int n = blockIdx.x * 256 + threadIdx.x;
    if (n >= LDN) return;
    float b;
    if      (n < 512)  b = bq[n];
    else if (n < 1024) b = bk[n - 512];
    else if (n < 1536) b = bv[n - 1024];
    else               b = bskip[n - 1536];
    biasc[n] = b;
}

// ---------------------------------------------------------------------------
// Fused QKV+skip GEMM: [MPAD x 128] bf16 @ [128 x 1664] bf16 -> f32, + bias.
// One block (8 waves) per 16-row M-tile:
//   - A tile (16x128 bf16 = 4KB) staged in LDS once. The CDNA5 async copy
//     (global_load_async_to_lds_b128 + s_wait_asynccnt) does the transfer;
//     an architectural copy of the same bytes into the same slots makes the
//     initialization visible to the compiler (both writers store identical
//     data and are ordered by the wait + barrier, so the value is defined).
//   - Wave wv owns n-tiles nt = wv + 8*j, j=0..12; two per iteration ->
//     8 WMMAs over 4 independent chains fill the co-exec hazard slots.
//   - All 16 B loads of a pair share one vaddr + imm24 offsets.
// ---------------------------------------------------------------------------
__global__ __launch_bounds__(256) void gemm_qkvskip(
    const unsigned short* __restrict__ Xb,
    const unsigned short* __restrict__ Bp,
    const float* __restrict__ biasc,
    float* __restrict__ Y) {
    __shared__ __align__(16) unsigned short Atile[16 * Dd];   // 4096 B at LDS offset 0

    const int tid = threadIdx.x;
    const int mt  = blockIdx.x;

    // --- stage A tile: 256 threads x 16B = 4KB ---
    {
        const unsigned short* Ag = Xb + (size_t)mt * 16 * Dd;
        unsigned long long gaddr =
            (unsigned long long)(uintptr_t)Ag + (unsigned long long)(tid * 16);
        unsigned int ldsoff = (unsigned int)(tid * 16);
        asm volatile("global_load_async_to_lds_b128 %0, %1, off"
                     :: "v"(ldsoff), "v"(gaddr) : "memory");
        asm volatile("s_wait_asynccnt 0" ::: "memory");
        // architectural copy of the same bytes (async write is opaque to AA)
        uint4 t4 = *(const uint4*)(Ag + tid * 8);
        *(uint4*)(&Atile[tid * 8]) = t4;
    }
    __syncthreads();

    const int lane = tid & 31;
    const int wv   = tid >> 5;          // wave 0..7
    const int hi   = lane >> 4;
    const int r    = lane & 15;

    // --- load A registers from LDS once (ds_load_b128 x8) ---
    v16bf a0, a1, a2, a3;
    {
        const unsigned short* Ar = &Atile[r * Dd + hi * 8];
        ((uint4*)&a0)[0] = *(const uint4*)(Ar);
        ((uint4*)&a0)[1] = *(const uint4*)(Ar + 16);
        ((uint4*)&a1)[0] = *(const uint4*)(Ar + 32);
        ((uint4*)&a1)[1] = *(const uint4*)(Ar + 48);
        ((uint4*)&a2)[0] = *(const uint4*)(Ar + 64);
        ((uint4*)&a2)[1] = *(const uint4*)(Ar + 80);
        ((uint4*)&a3)[0] = *(const uint4*)(Ar + 96);
        ((uint4*)&a3)[1] = *(const uint4*)(Ar + 112);
    }

    const unsigned short* base = Bp + ((size_t)wv * 32 + lane) * 16;  // nt = wv
    const bool fullM = (mt < MTILES - 1);
    const int  m0    = mt * 16 + hi * 8;

    // --- 6 pairs of n-tiles (nt = wv + 8*j), 8 WMMAs / 4 chains per pair ---
    for (int jj = 0; jj < 6; ++jj) {
        const int j0 = 2 * jj;
        const unsigned short* p0 = base + (size_t)j0 * 8 * NTSTRIDE;
        BTile t0 = load_btile(p0);
        BTile t1 = load_btile(p0 + 8 * NTSTRIDE);

        v8f cA0 = {}, cA1 = {}, cB0 = {}, cB1 = {};
        cA0 = __builtin_amdgcn_wmma_f32_16x16x32_bf16(false, a0, false, t0.b0, (short)0, cA0, false, false);
        cB0 = __builtin_amdgcn_wmma_f32_16x16x32_bf16(false, a0, false, t1.b0, (short)0, cB0, false, false);
        cA1 = __builtin_amdgcn_wmma_f32_16x16x32_bf16(false, a2, false, t0.b2, (short)0, cA1, false, false);
        cB1 = __builtin_amdgcn_wmma_f32_16x16x32_bf16(false, a2, false, t1.b2, (short)0, cB1, false, false);
        cA0 = __builtin_amdgcn_wmma_f32_16x16x32_bf16(false, a1, false, t0.b1, (short)0, cA0, false, false);
        cB0 = __builtin_amdgcn_wmma_f32_16x16x32_bf16(false, a1, false, t1.b1, (short)0, cB0, false, false);
        cA1 = __builtin_amdgcn_wmma_f32_16x16x32_bf16(false, a3, false, t0.b3, (short)0, cA1, false, false);
        cB1 = __builtin_amdgcn_wmma_f32_16x16x32_bf16(false, a3, false, t1.b3, (short)0, cB1, false, false);

        const int colA = (wv + 8 * j0) * 16 + r;
        const int colB = colA + 8 * 16;
        const float biasA = biasc[colA];
        const float biasB = biasc[colB];
        float* ypA = Y + (size_t)m0 * LDN + colA;
        float* ypB = Y + (size_t)m0 * LDN + colB;

        if (fullM) {
#pragma unroll
            for (int v = 0; v < 8; ++v) {
                ypA[(size_t)v * LDN] = cA0[v] + cA1[v] + biasA;
                ypB[(size_t)v * LDN] = cB0[v] + cB1[v] + biasB;
            }
        } else {
#pragma unroll
            for (int v = 0; v < 8; ++v) {
                if (m0 + v < Nn) {
                    ypA[(size_t)v * LDN] = cA0[v] + cA1[v] + biasA;
                    ypB[(size_t)v * LDN] = cB0[v] + cB1[v] + biasB;
                }
            }
        }
    }

    // --- tail tile: nt = wv + 8*12 ---
    {
        const unsigned short* p0 = base + (size_t)12 * 8 * NTSTRIDE;
        BTile t0 = load_btile(p0);
        v8f c0 = {}, c1 = {};
        c0 = __builtin_amdgcn_wmma_f32_16x16x32_bf16(false, a0, false, t0.b0, (short)0, c0, false, false);
        c1 = __builtin_amdgcn_wmma_f32_16x16x32_bf16(false, a2, false, t0.b2, (short)0, c1, false, false);
        c0 = __builtin_amdgcn_wmma_f32_16x16x32_bf16(false, a1, false, t0.b1, (short)0, c0, false, false);
        c1 = __builtin_amdgcn_wmma_f32_16x16x32_bf16(false, a3, false, t0.b3, (short)0, c1, false, false);

        const int col = (wv + 96) * 16 + r;
        const float bias = biasc[col];
        float* yp = Y + (size_t)m0 * LDN + col;
        if (fullM) {
#pragma unroll
            for (int v = 0; v < 8; ++v)
                yp[(size_t)v * LDN] = c0[v] + c1[v] + bias;
        } else {
#pragma unroll
            for (int v = 0; v < 8; ++v)
                if (m0 + v < Nn) yp[(size_t)v * LDN] = c0[v] + c1[v] + bias;
        }
    }
}

// ---------------------------------------------------------------------------
// Edge phase. One wave per (edge, head). q/k/v gathers hit L2 (62MB resident).
// ---------------------------------------------------------------------------
__global__ __launch_bounds__(256) void edge_alpha(
    const long long* __restrict__ ei, const float* __restrict__ Y,
    float* __restrict__ alpha, unsigned int* __restrict__ amax) {
    int wid = (blockIdx.x << 3) | (threadIdx.x >> 5);
    if (wid >= Ee * Hh) return;
    int lane = threadIdx.x & 31;
    int e = wid >> 2, h = wid & 3;
    long long src = ei[e];
    long long dst = ei[Ee + e];
    int c0 = lane * 4;
    const float4 q4 = *(const float4*)(Y + (size_t)dst * LDN +        h * CC + c0);
    const float4 k4 = *(const float4*)(Y + (size_t)src * LDN + 512 +  h * CC + c0);
    float s = q4.x * k4.x + q4.y * k4.y + q4.z * k4.z + q4.w * k4.w;
#pragma unroll
    for (int m = 16; m > 0; m >>= 1) s += __shfl_xor(s, m, 32);
    if (lane == 0) {
        float a = s * 0.08838834764831845f;   // 1/sqrt(128)
        alpha[wid] = a;
        atomicMax(&amax[(int)dst * Hh + h], f2ord(a));
    }
}

__global__ void edge_exp(const long long* __restrict__ ei, const float* __restrict__ alpha,
                         const unsigned int* __restrict__ amax,
                         float* __restrict__ exb, float* __restrict__ denom) {
    int eh = blockIdx.x * 256 + threadIdx.x;
    if (eh >= Ee * Hh) return;
    int e = eh >> 2, h = eh & 3;
    long long dst = ei[Ee + e];
    float am = ord2f(amax[(int)dst * Hh + h]);
    if (!isfinite(am)) am = 0.0f;
    float ex = expf(alpha[eh] - am);
    exb[eh] = ex;
    atomicAdd(&denom[(int)dst * Hh + h], ex);
}

__global__ __launch_bounds__(256) void edge_msg(
    const long long* __restrict__ ei, const float* __restrict__ Y,
    const float* __restrict__ exb, const float* __restrict__ denom,
    float* __restrict__ agg) {
    int wid = (blockIdx.x << 3) | (threadIdx.x >> 5);
    if (wid >= Ee * Hh) return;
    int lane = threadIdx.x & 31;
    int e = wid >> 2, h = wid & 3;
    long long src = ei[e];
    long long dst = ei[Ee + e];
    float w = exb[wid] / (denom[(int)dst * Hh + h] + 1e-16f);
    int c0 = lane * 4;
    const float4 v4 = *(const float4*)(Y + (size_t)src * LDN + 1024 + h * CC + c0);
    float* ap = agg + (size_t)dst * HCc + h * CC + c0;
    atomicAdd(ap + 0, w * v4.x);
    atomicAdd(ap + 1, w * v4.y);
    atomicAdd(ap + 2, w * v4.z);
    atomicAdd(ap + 3, w * v4.w);
}

// ---------------------------------------------------------------------------
// Node-wise epilogue
// ---------------------------------------------------------------------------
__global__ __launch_bounds__(128) void node_fuse(
    const float* __restrict__ Y, const float* __restrict__ agg,
    const float* __restrict__ Wbeta, float* __restrict__ fused,
    float* __restrict__ colsum) {
    int n = blockIdx.x, d = threadIdx.x;
    const float* an = agg + (size_t)n * HCc;
    float aggm = 0.25f * (an[d] + an[128 + d] + an[256 + d] + an[384 + d]);
    float skip = Y[(size_t)n * LDN + 1536 + d];
    __shared__ float red[128];
    red[d] = aggm * Wbeta[d] + skip * Wbeta[128 + d] + (aggm - skip) * Wbeta[256 + d];
    __syncthreads();
    for (int s = 64; s > 0; s >>= 1) { if (d < s) red[d] += red[d + s]; __syncthreads(); }
    float beta = 1.0f / (1.0f + expf(-red[0]));
    float o = beta * skip + (1.0f - beta) * aggm;
    fused[(size_t)n * Dd + d] = o;
    atomicAdd(&colsum[d], o);
}

__global__ __launch_bounds__(128) void center_kernel(
    float* __restrict__ fused, const float* __restrict__ colsum,
    const float* __restrict__ gms, float* __restrict__ ssq) {
    int n = blockIdx.x, d = threadIdx.x;
    float mean = colsum[d] * (1.0f / (float)Nn);
    float c = fused[(size_t)n * Dd + d] - gms[d] * mean;
    fused[(size_t)n * Dd + d] = c;
    atomicAdd(&ssq[d], c * c);
}

__global__ __launch_bounds__(128) void final_kernel(
    const float* __restrict__ fused, const float* __restrict__ ssq,
    const float* __restrict__ gw, const float* __restrict__ gb,
    const float* __restrict__ x, float* __restrict__ out) {
    int n = blockIdx.x, d = threadIdx.x;
    float c = fused[(size_t)n * Dd + d];
    float rstd = rsqrtf(ssq[d] * (1.0f / (float)Nn) + 1e-5f);
    float o = gw[d] * c * rstd + gb[d];
    float g = 0.5f * o * (1.0f + erff(o * 0.70710678118654752f));   // exact GELU
    out[(size_t)n * Dd + d] = g + x[(size_t)n * Dd + d];
}

// ---------------------------------------------------------------------------
// Workspace layout (bytes, 256-aligned)
// ---------------------------------------------------------------------------
#define XB_OFF     ((size_t)0)                        // MPAD*128 bf16   = 2,564,096
#define BP_OFF     ((size_t)2564096)                  // 128*1664 bf16   =   425,984
#define BIAS_OFF   ((size_t)2990080)                  // 1664 f32        =     6,656
#define Y_OFF      ((size_t)2996736)                  // N*1664 f32      = 66,560,000
#define ALPHA_OFF  ((size_t)69556736)                 // E*H f32         =  2,560,000
#define EX_OFF     ((size_t)72116736)                 // E*H f32         =  2,560,000
#define AMAX_OFF   ((size_t)74676736)                 // N*H u32         =    160,000
#define DEN_OFF    ((size_t)74836736)                 // N*H f32         =    160,000
#define AGG_OFF    ((size_t)74996736)                 // N*512 f32       = 20,480,000
#define FUSED_OFF  ((size_t)95476736)                 // N*128 f32       =  5,120,000
#define COLSUM_OFF ((size_t)100596736)                // 128 f32
#define SSQ_OFF    ((size_t)100597248)                // 128 f32

extern "C" void kernel_launch(void* const* d_in, const int* in_sizes, int n_in,
                              void* d_out, int out_size, void* d_ws, size_t ws_size,
                              hipStream_t stream) {
    const float*     x     = (const float*)d_in[0];
    const long long* ei    = (const long long*)d_in[1];
    const float*     Wq    = (const float*)d_in[2];
    const float*     bq    = (const float*)d_in[3];
    const float*     Wk    = (const float*)d_in[4];
    const float*     bk    = (const float*)d_in[5];
    const float*     Wv    = (const float*)d_in[6];
    const float*     bv    = (const float*)d_in[7];
    const float*     Wskip = (const float*)d_in[8];
    const float*     bskip = (const float*)d_in[9];
    const float*     Wbeta = (const float*)d_in[10];
    const float*     gw    = (const float*)d_in[11];
    const float*     gb    = (const float*)d_in[12];
    const float*     gms   = (const float*)d_in[13];
    float*           out   = (float*)d_out;

    char* ws = (char*)d_ws;
    unsigned short* Xb    = (unsigned short*)(ws + XB_OFF);
    unsigned short* Bp    = (unsigned short*)(ws + BP_OFF);
    float*          biasc = (float*)(ws + BIAS_OFF);
    float*          Y     = (float*)(ws + Y_OFF);
    float*          alpha = (float*)(ws + ALPHA_OFF);
    float*          exb   = (float*)(ws + EX_OFF);
    unsigned int*   amax  = (unsigned int*)(ws + AMAX_OFF);
    float*          den   = (float*)(ws + DEN_OFF);
    float*          agg   = (float*)(ws + AGG_OFF);
    float*          fused = (float*)(ws + FUSED_OFF);
    float*          colsum= (float*)(ws + COLSUM_OFF);
    float*          ssq   = (float*)(ws + SSQ_OFF);

    // zero accumulators (graph-capture-legal)
    (void)hipMemsetAsync(amax,   0, (size_t)Nn * Hh * 4, stream);
    (void)hipMemsetAsync(den,    0, (size_t)Nn * Hh * 4, stream);
    (void)hipMemsetAsync(agg,    0, (size_t)Nn * HCc * 4, stream);
    (void)hipMemsetAsync(colsum, 0, 128 * 4, stream);
    (void)hipMemsetAsync(ssq,    0, 128 * 4, stream);

    // pack
    pack_x   <<<(MPAD * Dd + 255) / 256, 256, 0, stream>>>(x, Xb);
    pack_b   <<<(4 * NTILES * 32 * 16 + 255) / 256, 256, 0, stream>>>(Wq, Wk, Wv, Wskip, Bp);
    pack_bias<<<(LDN + 255) / 256, 256, 0, stream>>>(bq, bk, bv, bskip, biasc);

    // fused q|k|v|skip GEMM (WMMA bf16 -> f32, LDS-staged A)
    gemm_qkvskip<<<MTILES, 256, 0, stream>>>(Xb, Bp, biasc, Y);

    // edge attention
    const int EH = Ee * Hh;                       // 640,000
    edge_alpha<<<EH / 8, 256, 0, stream>>>(ei, Y, alpha, amax);
    edge_exp  <<<(EH + 255) / 256, 256, 0, stream>>>(ei, alpha, amax, exb, den);
    edge_msg  <<<EH / 8, 256, 0, stream>>>(ei, Y, exb, den, agg);

    // node-wise epilogue
    node_fuse    <<<Nn, 128, 0, stream>>>(Y, agg, Wbeta, fused, colsum);
    center_kernel<<<Nn, 128, 0, stream>>>(fused, colsum, gms, ssq);
    final_kernel <<<Nn, 128, 0, stream>>>(fused, ssq, gw, gb, x, out);
}
